// TurboQuantEstimator_27075473834425
// MI455X (gfx1250) — matrix operand order repaired and checked by reading, not compiled, not run on hardware
//
#include <hip/hip_runtime.h>
#include <hip/hip_bf16.h>
#include <math.h>

typedef __attribute__((ext_vector_type(16))) _Float16 v16h;
typedef __attribute__((ext_vector_type(8)))  _Float16 v8h;
typedef __attribute__((ext_vector_type(8)))  float    v8f;

#define D 128            // feature dim
#define M 128            // projection dim
#define KC 256           // fused GEMM K = D + M
#define EPSF 1e-8f
#define QJL_SCALE 0.011048543456039806f   // sqrt(2)/128

// ---------------------------------------------------------------------------
// A-operand fragment: 16x32 fp16 tile (M x K).  ISA layout:
// lane l holds row m = l%16; K chunks (l/16)*8 + {0..7} in V0-3 and
// (l/16)*8 + 16 + {0..7} in V4-7.  base -> (row0,k0); stride in halves.
// ---------------------------------------------------------------------------
__device__ __forceinline__ v16h load_frag_a(const _Float16* base, int stride, int lane) {
    const _Float16* p = base + (size_t)(lane & 15) * stride + ((lane >> 4) << 3);
    v8h lo = *(const v8h*)(p);
    v8h hi = *(const v8h*)(p + 16);
    return __builtin_shufflevector(lo, hi, 0,1,2,3,4,5,6,7,8,9,10,11,12,13,14,15);
}

// ---------------------------------------------------------------------------
// B-operand fragment: 32x16 fp16 tile (K x N), source stored N-major
// (row n contiguous in k, i.e. B^T row-major).  ISA layout: lane l holds
// column n = l%16; K chunks (l/16)*8 + {0..7} in V0-3 and
// (l/16)*8 + 16 + {0..7} in V4-7 (same k order as A).
// ---------------------------------------------------------------------------
__device__ __forceinline__ v16h load_frag_b(const _Float16* base, int stride, int lane) {
    const _Float16* p = base + (size_t)(lane & 15) * stride + ((lane >> 4) << 3);
    v8h lo = *(const v8h*)(p);
    v8h hi = *(const v8h*)(p + 16);
    return __builtin_shufflevector(lo, hi, 0,1,2,3,4,5,6,7,8,9,10,11,12,13,14,15);
}

// ---------------------------------------------------------------------------
// Kernel 0: convert S (m x d fp32) -> fp16 copy (row major)
// ---------------------------------------------------------------------------
__global__ void s_to_fp16(const float* __restrict__ S, _Float16* __restrict__ S16) {
    int i = blockIdx.x * blockDim.x + threadIdx.x;
    if (i < M * D) S16[i] = (_Float16)S[i];
}

// ---------------------------------------------------------------------------
// Kernel 1: A panel.  Aq[q][0:128] = fp16(query[q]); Aq[q][128:256] = fp16(q @ S^T)
// one block per query row, 128 threads.
// ---------------------------------------------------------------------------
__global__ void a_prep(const float* __restrict__ query, const float* __restrict__ S,
                       _Float16* __restrict__ Aq) {
    __shared__ float qrow[D];
    const int t = threadIdx.x;
    const int q = blockIdx.x;
    float v = query[(size_t)q * D + t];
    qrow[t] = v;
    Aq[(size_t)q * KC + t] = (_Float16)v;
    __syncthreads();
    const float* srow = S + (size_t)t * D;   // S row t, contiguous
    float acc = 0.f;
#pragma unroll 4
    for (int j = 0; j < D; j += 4) {
        acc = fmaf(qrow[j+0], srow[j+0], acc);
        acc = fmaf(qrow[j+1], srow[j+1], acc);
        acc = fmaf(qrow[j+2], srow[j+2], acc);
        acc = fmaf(qrow[j+3], srow[j+3], acc);
    }
    Aq[(size_t)q * KC + D + t] = (_Float16)acc;
}

// ---------------------------------------------------------------------------
// Kernel 2: B panel.  16 keys per block, 128 threads (4 waves).
// Phase 1 (VALU): norms, 4-level codebook quantize, residual -> LDS (fp16),
//                 Bk[n][0:128] = fp16(x_mse * ||k||).
// Phase 2 (WMMA): P = R(16x128) @ S^T(128x128); Bk[n][128+i] = sign(P)*g[n]
//                 with g = r_norm * sqrt(2)/m * ||k||  (vec_norm folded in).
// ---------------------------------------------------------------------------
__global__ void b_prep(const float* __restrict__ keys, const float* __restrict__ codebook,
                       const _Float16* __restrict__ S16, _Float16* __restrict__ Bk) {
    __shared__ _Float16 Rl[16][136];   // residual tile, padded row stride
    __shared__ float    gsh[16];

    const int tid = threadIdx.x;
    const int k   = tid >> 3;          // key within block: 0..15
    const int w   = tid & 7;           // 8 lanes per key, 16 coords each
    const int n   = blockIdx.x * 16 + k;

    const float c0 = codebook[0], c1 = codebook[1], c2 = codebook[2], c3 = codebook[3];

    const float* kp = keys + (size_t)n * D + w * 16;
    float x[16];
#pragma unroll
    for (int i = 0; i < 16; i += 4) {
        float4 t4 = *(const float4*)(kp + i);
        x[i] = t4.x; x[i+1] = t4.y; x[i+2] = t4.z; x[i+3] = t4.w;
    }
    float ss = 0.f;
#pragma unroll
    for (int i = 0; i < 16; ++i) ss = fmaf(x[i], x[i], ss);
    ss += __shfl_xor(ss, 1); ss += __shfl_xor(ss, 2); ss += __shfl_xor(ss, 4);
    const float vnorm = sqrtf(ss);
    const float inv   = 1.f / (vnorm + EPSF);

    float rr = 0.f;
    _Float16 bv[16];
#pragma unroll
    for (int i = 0; i < 16; ++i) {
        float xn = x[i] * inv;
        // nearest of 4 sorted centers
        float xm = c0, best = fabsf(xn - c0), dd;
        dd = fabsf(xn - c1); if (dd < best) { best = dd; xm = c1; }
        dd = fabsf(xn - c2); if (dd < best) { best = dd; xm = c2; }
        dd = fabsf(xn - c3); if (dd < best) { best = dd; xm = c3; }
        float rv = xn - xm;
        rr = fmaf(rv, rv, rr);
        Rl[k][w * 16 + i] = (_Float16)rv;
        bv[i] = (_Float16)(xm * vnorm);          // fold ||k|| into B
    }
    { // Bk[n][0:128]
        _Float16* bp = Bk + (size_t)n * KC + w * 16;
        *(v8h*)(bp)     = *(v8h*)&bv[0];
        *(v8h*)(bp + 8) = *(v8h*)&bv[8];
    }
    rr += __shfl_xor(rr, 1); rr += __shfl_xor(rr, 2); rr += __shfl_xor(rr, 4);
    if (w == 0) gsh[k] = sqrtf(rr) * QJL_SCALE * vnorm;   // fold ||k|| into g
    __syncthreads();

    // Phase 2: WMMA projection. 4 waves x 2 proj-tiles of 16 each = 128 projections.
    const int lane = tid & 31;
    const int wv   = tid >> 5;
#pragma unroll
    for (int t = 0; t < 2; ++t) {
        const int tile = wv * 2 + t;          // proj tile 0..7
        v8f acc = {};
#pragma unroll
        for (int ks = 0; ks < 4; ++ks) {
            // A: residual from LDS (A-operand layout)
            const _Float16* ap = &Rl[lane & 15][ks * 32 + ((lane >> 4) << 3)];
            v8h alo = *(const v8h*)(ap);
            v8h ahi = *(const v8h*)(ap + 16);
            v16h a = __builtin_shufflevector(alo, ahi, 0,1,2,3,4,5,6,7,8,9,10,11,12,13,14,15);
            // B: S^T column block = S16 rows (tile*16 .. tile*16+15), B-operand layout
            v16h b = load_frag_b(S16 + (size_t)(tile * 16) * D + ks * 32, D, lane);
            acc = __builtin_amdgcn_wmma_f32_16x16x32_f16(false, a, false, b,
                                                         (short)0, acc, false, false);
        }
#pragma unroll
        for (int r = 0; r < 8; ++r) {
            int key = r + ((lane >> 4) << 3);
            int pi  = tile * 16 + (lane & 15);
            float g = gsh[key];
            float v = (acc[r] >= 0.f) ? g : -g;   // sign (>=0 -> +1), scaled
            Bk[(size_t)(blockIdx.x * 16 + key) * KC + D + pi] = (_Float16)v;
        }
    }
}

// ---------------------------------------------------------------------------
// Kernel 3: fused GEMM  out(Q x N) = Aq(Q x 256) @ Bk(N x 256)^T
// block = 256 threads (8 waves); each wave owns a 64x64 tile; block covers
// 64 q-rows x 512 n-cols.  8 K-steps of 16 wmma_f32_16x16x32_f16.
// ---------------------------------------------------------------------------
__global__ void __launch_bounds__(256)
fused_gemm(const _Float16* __restrict__ Aq, const _Float16* __restrict__ Bk,
           float* __restrict__ out, int N) {
    const int lane  = threadIdx.x & 31;
    const int wv    = threadIdx.x >> 5;
    const int qbase = blockIdx.y * 64;
    const int nbase = blockIdx.x * 512 + wv * 64;

    v8f acc[4][4];
#pragma unroll
    for (int mi = 0; mi < 4; ++mi)
#pragma unroll
        for (int ni = 0; ni < 4; ++ni) acc[mi][ni] = (v8f){};

#pragma unroll
    for (int ks = 0; ks < 8; ++ks) {
        const int kb = ks * 32;
        v16h a[4], b[4];
#pragma unroll
        for (int mi = 0; mi < 4; ++mi)
            a[mi] = load_frag_a(Aq + (size_t)(qbase + mi * 16) * KC + kb, KC, lane);
#pragma unroll
        for (int ni = 0; ni < 4; ++ni)
            b[ni] = load_frag_b(Bk + (size_t)(nbase + ni * 16) * KC + kb, KC, lane);
#pragma unroll
        for (int mi = 0; mi < 4; ++mi)
#pragma unroll
            for (int ni = 0; ni < 4; ++ni)
                acc[mi][ni] = __builtin_amdgcn_wmma_f32_16x16x32_f16(
                    false, a[mi], false, b[ni], (short)0, acc[mi][ni], false, false);
    }

    // C/D layout: n = lane & 15; M = vgpr r + 8*(lane/16)
#pragma unroll
    for (int mi = 0; mi < 4; ++mi) {
#pragma unroll
        for (int ni = 0; ni < 4; ++ni) {
            const int col = nbase + ni * 16 + (lane & 15);
#pragma unroll
            for (int r = 0; r < 8; ++r) {
                const int row = qbase + mi * 16 + ((lane >> 4) << 3) + r;
                out[(size_t)row * N + col] = acc[mi][ni][r];
            }
        }
    }
}

// ---------------------------------------------------------------------------
extern "C" void kernel_launch(void* const* d_in, const int* in_sizes, int n_in,
                              void* d_out, int out_size, void* d_ws, size_t ws_size,
                              hipStream_t stream) {
    const float* query    = (const float*)d_in[0];
    const float* keys     = (const float*)d_in[1];
    const float* codebook = (const float*)d_in[2];
    const float* S        = (const float*)d_in[3];
    float* out = (float*)d_out;

    const int Q = in_sizes[0] / D;   // 512
    const int N = in_sizes[1] / D;   // 131072

    // workspace layout: Aq (Q*256 fp16) | S16 (128*128 fp16) | Bk (N*256 fp16)
    char* ws = (char*)d_ws;
    _Float16* Aq  = (_Float16*)ws;
    _Float16* S16 = (_Float16*)(ws + (size_t)Q * KC * sizeof(_Float16));
    _Float16* Bk  = (_Float16*)(ws + (size_t)Q * KC * sizeof(_Float16)
                                   + (size_t)M * D * sizeof(_Float16));

    s_to_fp16<<<(M * D + 255) / 256, 256, 0, stream>>>(S, S16);
    a_prep<<<Q, 128, 0, stream>>>(query, S, Aq);
    b_prep<<<N / 16, 128, 0, stream>>>(keys, codebook, S16, Bk);
    fused_gemm<<<dim3(N / 512, Q / 64), 256, 0, stream>>>(Aq, Bk, out, N);
}
